// FastTransformerWrapper_3478923509951
// MI455X (gfx1250) — compile-verified, matrix operand order checked by reading, not run on hardware
//
#include <hip/hip_runtime.h>
#include <hip/hip_bf16.h>
#include <math.h>

// ---------------------------------------------------------------------------
// fast_transformers linear-attention encoder on MI455X (gfx1250, wave32).
// GEMMs: v_wmma_f32_16x16x32_bf16, f32 accumulate, double-buffered LDS fed by
// GLOBAL_LOAD_ASYNC_TO_LDS_B128 (ASYNCcnt), one barrier per k-step.
// Weights pre-converted+transposed to bf16 once per launch; activations carry
// bf16 shadows.  Attention einsums / LayerNorm: VALU f32 (~2% of FLOPs).
// ---------------------------------------------------------------------------

typedef __bf16 bf16_t;
typedef bf16_t v16bf __attribute__((ext_vector_type(16)));
typedef bf16_t v8bf  __attribute__((ext_vector_type(8)));
typedef bf16_t v2bf  __attribute__((ext_vector_type(2)));
typedef float  v8f   __attribute__((ext_vector_type(8)));

#define NB     4
#define LSEQ   8192
#define DMODEL 512
#define NHEAD  8
#define DHEAD  64
#define DFFN   2048
#define NLAYER 4
#define ROWS   (NB * LSEQ)   // 32768

enum EpiMode { EPI_BIAS = 0, EPI_ELU1 = 1, EPI_ELU1_MASK = 2, EPI_GELU = 3, EPI_RESID = 4 };

__device__ __forceinline__ v8f wmma_bf16(v16bf a, v16bf b, v8f c) {
  return __builtin_amdgcn_wmma_f32_16x16x32_bf16(false, a, false, b, (short)0, c, false, false);
}

// 16-bit A/B fragment (ISA 7.12.2): lane holds row/col (lane&15); elements
// 0..7 = K in [hk,hk+8), 8..15 = K in [16+hk,16+hk+8), hk=(lane>=16)*8.
__device__ __forceinline__ v16bf load_frag(const bf16_t* rowptr, int hk) {
  v8bf lo = *(const v8bf*)(rowptr + hk);
  v8bf hi = *(const v8bf*)(rowptr + 16 + hk);
  return __builtin_shufflevector(lo, hi, 0, 1, 2, 3, 4, 5, 6, 7, 8, 9, 10, 11, 12, 13, 14, 15);
}

// Async global->LDS 16B/lane copy (VGLOBAL op 98, tracked by ASYNCcnt).
// lds_addr = per-lane LDS byte offset (low 32 bits of generic LDS pointer).
__device__ __forceinline__ void async_copy_b128(unsigned lds_addr, const bf16_t* gptr) {
  asm volatile("global_load_async_to_lds_b128 %0, %1, off"
               :: "v"(lds_addr), "v"(gptr) : "memory");
}

__device__ __forceinline__ void wait_async0() {
#if __has_builtin(__builtin_amdgcn_s_wait_asynccnt)
  __builtin_amdgcn_s_wait_asynccnt(0);
#else
  asm volatile("s_wait_asynccnt 0x0" ::: "memory");
#endif
}

// ---------------------------------------------------------------------------
// GEMM: out[M,Nc] = epi(A[M,K](bf16) @ Bt[Nc,K](bf16, pre-transposed) + bias)
// Block 128x128, 8 waves each 32x64 (2x4 WMMA tiles).  K,Nc compile-time.
// ---------------------------------------------------------------------------
template <int MODE, int K, int Nc>
__global__ __launch_bounds__(256)
void gemm_bf16_kernel(const bf16_t* __restrict__ A, const bf16_t* __restrict__ Bt,
                      const float* __restrict__ bias, const float* __restrict__ resid,
                      const int* __restrict__ lengths,
                      float* __restrict__ outf, bf16_t* __restrict__ outh) {
  __shared__ bf16_t As[2][128][32];   // 8 KB per buffer
  __shared__ bf16_t Bs[2][128][32];

  const int tid  = threadIdx.x;
  const int lane = tid & 31;
  const int wid  = tid >> 5;
  const int wm   = wid >> 1;            // 0..3 -> 32-row strip
  const int wn   = wid & 1;             // 0..1 -> 64-col strip
  const int m0   = blockIdx.y * 128;
  const int n0   = blockIdx.x * 128;
  const int r15  = lane & 15;
  const int hk   = (lane >> 4) << 3;

  // staging: thread owns rows lr, lr+64 and 8 consecutive k -> one b128 each
  const int lr = tid >> 2;              // 0..63
  const int lc = (tid & 3) * 8;         // 0,8,16,24
  const bf16_t* Ap = A  + (size_t)(m0 + lr) * K + lc;
  const bf16_t* Bp = Bt + (size_t)(n0 + lr) * K + lc;
  const size_t row64 = (size_t)64 * K;

  const unsigned ldsA = (unsigned)(uintptr_t)&As[0][lr][lc];  // LDS byte offset
  const unsigned ldsB = (unsigned)(uintptr_t)&Bs[0][lr][lc];
  const unsigned BUFS = 128 * 32 * sizeof(bf16_t);            // 8192
  const unsigned ROWS64 = 64 * 32 * sizeof(bf16_t);           // 4096

  v8f acc[2][4];
#pragma unroll
  for (int i = 0; i < 2; i++)
#pragma unroll
    for (int j = 0; j < 4; j++) acc[i][j] = (v8f){0.f, 0.f, 0.f, 0.f, 0.f, 0.f, 0.f, 0.f};

  // tile 0 -> buffer 0
  async_copy_b128(ldsA,          Ap);
  async_copy_b128(ldsA + ROWS64, Ap + row64);
  async_copy_b128(ldsB,          Bp);
  async_copy_b128(ldsB + ROWS64, Bp + row64);
  wait_async0();
  __syncthreads();

  int buf = 0;
#pragma unroll 2
  for (int kb = 0; kb < K; kb += 32, buf ^= 1) {
    const bool more = (kb + 32) < K;
    if (more) {                        // async-prefetch next tile during WMMA
      const unsigned nb = (buf ^ 1) * BUFS;
      async_copy_b128(ldsA + nb,          Ap + kb + 32);
      async_copy_b128(ldsA + nb + ROWS64, Ap + row64 + kb + 32);
      async_copy_b128(ldsB + nb,          Bp + kb + 32);
      async_copy_b128(ldsB + nb + ROWS64, Bp + row64 + kb + 32);
    }
    v16bf af[2], bfm[4];
#pragma unroll
    for (int mt = 0; mt < 2; mt++) af[mt] = load_frag(&As[buf][wm * 32 + mt * 16 + r15][0], hk);
#pragma unroll
    for (int nt = 0; nt < 4; nt++) bfm[nt] = load_frag(&Bs[buf][wn * 64 + nt * 16 + r15][0], hk);
#pragma unroll
    for (int mt = 0; mt < 2; mt++)
#pragma unroll
      for (int nt = 0; nt < 4; nt++) acc[mt][nt] = wmma_bf16(af[mt], bfm[nt], acc[mt][nt]);
    if (more) {
      wait_async0();                   // async writes landed in LDS
      __syncthreads();                 // single barrier per k-step
    }
  }

  // C/D layout: lane<16 -> M rows 0..7, N=lane; lane>=16 -> M rows 8..15.
  const int mrow = (lane >> 4) << 3;
#pragma unroll
  for (int mt = 0; mt < 2; mt++) {
    const int rowb = m0 + wm * 32 + mt * 16 + mrow;
#pragma unroll
    for (int nt = 0; nt < 4; nt++) {
      const int col = n0 + wn * 64 + nt * 16 + r15;
      const float bv = bias[col];
      const size_t base = (size_t)rowb * Nc + col;
#pragma unroll
      for (int v = 0; v < 8; v++) {
        float x = acc[mt][nt][v] + bv;
        if constexpr (MODE == EPI_ELU1) {
          x = x > 0.f ? x + 1.f : __expf(x);
        } else if constexpr (MODE == EPI_ELU1_MASK) {
          x = x > 0.f ? x + 1.f : __expf(x);
          const int row = rowb + v;
          x = ((row & (LSEQ - 1)) < lengths[row >> 13]) ? x : 0.f;
        } else if constexpr (MODE == EPI_RESID) {
          x += resid[base + (size_t)v * Nc];
        }
        if constexpr (MODE == EPI_GELU) {
          float g = 0.5f * x * (1.f + erff(x * 0.70710678118654752f));
          outh[base + (size_t)v * Nc] = (bf16_t)g;   // bf16-only (FFN hidden)
        } else {
          outf[base + (size_t)v * Nc] = x;
        }
      }
    }
  }
}

// ---------------------------------------------------------------------------
// one-time weight convert+transpose: WT[n][k] = (bf16)W[k][n]
// ---------------------------------------------------------------------------
__global__ __launch_bounds__(256)
void convert_transpose_kernel(const float* __restrict__ W, bf16_t* __restrict__ WT,
                              int K, int N) {
  __shared__ float t[32][33];
  const int k0 = blockIdx.y * 32, n0 = blockIdx.x * 32;
  const int tx = threadIdx.x & 31, ty = threadIdx.x >> 5;
  for (int i = ty; i < 32; i += 8) t[i][tx] = W[(size_t)(k0 + i) * N + n0 + tx];
  __syncthreads();
  for (int i = ty; i < 32; i += 8) WT[(size_t)(n0 + i) * K + k0 + tx] = (bf16_t)t[tx][i];
}

__global__ __launch_bounds__(256)
void f32_to_bf16_kernel(const float* __restrict__ in, bf16_t* __restrict__ out, int n) {
  int i = blockIdx.x * 256 + threadIdx.x;
  if (i < n) out[i] = (bf16_t)in[i];
}

__global__ __launch_bounds__(256)
void lengths_kernel(const unsigned char* __restrict__ mask, int* __restrict__ lengths) {
  int n = blockIdx.x, t = threadIdx.x;
  int s = 0;
  for (int i = t; i < LSEQ; i += 256) s += mask[n * LSEQ + i] ? 1 : 0;
  __shared__ int red[256];
  red[t] = s;
  __syncthreads();
  for (int o = 128; o >= 1; o >>= 1) {
    if (t < o) red[t] += red[t + o];
    __syncthreads();
  }
  if (t == 0) lengths[n] = LSEQ - red[0];
}

__global__ __launch_bounds__(256)
void zero_kernel(float* __restrict__ p, int n) {
  int i = blockIdx.x * 256 + threadIdx.x;
  if (i < n) p[i] = 0.f;
}

// ---------------------------------------------------------------------------
// KV[n,h,d,m] = sum_l K[n,l,h,d]*V[n,l,h,m];  Ksum[n,h,d] = sum_l K[n,l,h,d]
// ---------------------------------------------------------------------------
__global__ __launch_bounds__(256)
void kv_reduce_kernel(const float* __restrict__ Kp, const float* __restrict__ Vb,
                      float* __restrict__ KV, float* __restrict__ Ksum) {
  __shared__ float Kt[64][DHEAD + 1];
  __shared__ float Vt[64][DHEAD + 1];
  const int nh = blockIdx.x;
  const int n = nh >> 3, h = nh & 7;
  const int tid = threadIdx.x, lane = tid & 31, w = tid >> 5;
  const int Lc = LSEQ / gridDim.y;
  const int lbase = blockIdx.y * Lc;
  const int m0 = lane * 2;

  float acc[8][2];
#pragma unroll
  for (int i = 0; i < 8; i++) { acc[i][0] = 0.f; acc[i][1] = 0.f; }
  float ks0 = 0.f, ks1 = 0.f;

  for (int lcofs = 0; lcofs < Lc; lcofs += 64) {
    __syncthreads();
#pragma unroll
    for (int j = 0; j < 16; j++) {
      int idx = tid + 256 * j;
      int l = idx >> 6, d = idx & 63;
      size_t base = ((size_t)(n * LSEQ + lbase + lcofs + l)) * DMODEL + h * DHEAD + d;
      Kt[l][d] = Kp[base];
      Vt[l][d] = Vb[base];
    }
    __syncthreads();
#pragma unroll 4
    for (int l = 0; l < 64; l++) {
      float v0 = Vt[l][m0], v1 = Vt[l][m0 + 1];
#pragma unroll
      for (int dd = 0; dd < 8; dd++) {
        float kv = Kt[l][w * 8 + dd];
        acc[dd][0] += kv * v0;
        acc[dd][1] += kv * v1;
      }
    }
    if (w == 0) {
#pragma unroll 4
      for (int l = 0; l < 64; l++) { ks0 += Kt[l][m0]; ks1 += Kt[l][m0 + 1]; }
    }
  }
  size_t kvbase = (size_t)nh * DHEAD * DHEAD;
#pragma unroll
  for (int dd = 0; dd < 8; dd++) {
    atomicAdd(&KV[kvbase + (size_t)(w * 8 + dd) * DHEAD + m0], acc[dd][0]);
    atomicAdd(&KV[kvbase + (size_t)(w * 8 + dd) * DHEAD + m0 + 1], acc[dd][1]);
  }
  if (w == 0) {
    atomicAdd(&Ksum[nh * DHEAD + m0], ks0);
    atomicAdd(&Ksum[nh * DHEAD + m0 + 1], ks1);
  }
}

// ---------------------------------------------------------------------------
// attn[n,l,h,m] = (sum_d Q*KV) / (Q.Ksum + eps), written as bf16 (next GEMM A)
// ---------------------------------------------------------------------------
__global__ __launch_bounds__(256)
void attn_apply_kernel(const float* __restrict__ Qp, const float* __restrict__ KV,
                       const float* __restrict__ Ksum, bf16_t* __restrict__ Ab) {
  __shared__ float KVs[DHEAD][DHEAD + 1];
  __shared__ float Kss[DHEAD];
  const int nh = blockIdx.x;
  const int n = nh >> 3, h = nh & 7;
  const int tid = threadIdx.x, lane = tid & 31, w = tid >> 5;
#pragma unroll
  for (int j = 0; j < 16; j++) {
    int idx = tid + 256 * j;
    KVs[idx >> 6][idx & 63] = KV[(size_t)nh * DHEAD * DHEAD + idx];
  }
  if (tid < DHEAD) Kss[tid] = Ksum[nh * DHEAD + tid];
  __syncthreads();

  const int m0 = lane * 2;
  for (int l = blockIdx.y * 8 + w; l < LSEQ; l += gridDim.y * 8) {
    size_t base = ((size_t)(n * LSEQ + l)) * DMODEL + h * DHEAD;
    float q0 = Qp[base + m0], q1 = Qp[base + m0 + 1];
    float s = q0 * Kss[m0] + q1 * Kss[m0 + 1];
#pragma unroll
    for (int o = 16; o >= 1; o >>= 1) s += __shfl_xor(s, o, 32);
    float Z = 1.f / (s + 1e-6f);
    float o0 = 0.f, o1 = 0.f;
#pragma unroll
    for (int d = 0; d < DHEAD; d++) {
      float qd = __shfl((d & 1) ? q1 : q0, d >> 1, 32);
      o0 += qd * KVs[d][m0];
      o1 += qd * KVs[d][m0 + 1];
    }
    *(v2bf*)&Ab[base + m0] = (v2bf){(bf16_t)(o0 * Z), (bf16_t)(o1 * Z)};
  }
}

// ---------------------------------------------------------------------------
// LayerNorm (rows of 512), wave per row; writes f32 x and optional bf16 shadow
// ---------------------------------------------------------------------------
__global__ __launch_bounds__(256)
void ln_kernel(const float* __restrict__ y, float* __restrict__ xo, bf16_t* __restrict__ xh,
               const float* __restrict__ g, const float* __restrict__ b, int rows) {
  const int w = threadIdx.x >> 5, lane = threadIdx.x & 31;
  const int r = blockIdx.x * 8 + w;
  if (r >= rows) return;
  const float* yr = y + (size_t)r * DMODEL;
  float e[16];
  float s = 0.f;
#pragma unroll
  for (int i = 0; i < 16; i++) { e[i] = yr[i * 32 + lane]; s += e[i]; }
#pragma unroll
  for (int o = 16; o >= 1; o >>= 1) s += __shfl_xor(s, o, 32);
  float mean = s * (1.f / 512.f);
  float vs = 0.f;
#pragma unroll
  for (int i = 0; i < 16; i++) { float d = e[i] - mean; vs += d * d; }
#pragma unroll
  for (int o = 16; o >= 1; o >>= 1) vs += __shfl_xor(vs, o, 32);
  float inv = rsqrtf(vs * (1.f / 512.f) + 1e-5f);
  float* xr = xo + (size_t)r * DMODEL;
#pragma unroll
  for (int i = 0; i < 16; i++) {
    int c = i * 32 + lane;
    float val = (e[i] - mean) * inv * g[c] + b[c];
    xr[c] = val;
    if (xh) xh[(size_t)r * DMODEL + c] = (bf16_t)val;
  }
}

// ---------------------------------------------------------------------------
extern "C" void kernel_launch(void* const* d_in, const int* in_sizes, int n_in,
                              void* d_out, int out_size, void* d_ws, size_t ws_size,
                              hipStream_t stream) {
  const float*         src  = (const float*)d_in[0];
  const unsigned char* mask = (const unsigned char*)d_in[1];
  const float* Wq = (const float*)d_in[2];  const float* bq = (const float*)d_in[3];
  const float* Wk = (const float*)d_in[4];  const float* bk = (const float*)d_in[5];
  const float* Wv = (const float*)d_in[6];  const float* bv = (const float*)d_in[7];
  const float* Wo = (const float*)d_in[8];  const float* bo = (const float*)d_in[9];
  const float* W1 = (const float*)d_in[10]; const float* b1 = (const float*)d_in[11];
  const float* W2 = (const float*)d_in[12]; const float* b2 = (const float*)d_in[13];
  const float* g1 = (const float*)d_in[14]; const float* be1 = (const float*)d_in[15];
  const float* g2 = (const float*)d_in[16]; const float* be2 = (const float*)d_in[17];
  const float* gf = (const float*)d_in[18]; const float* bef = (const float*)d_in[19];

  char* p = (char*)d_ws;
  auto alloc = [&](size_t bytes) -> void* {
    void* r = (void*)p;
    p += (bytes + 255) & ~(size_t)255;
    return r;
  };
  const size_t NLD = (size_t)ROWS * DMODEL;
  float*  KVb  = (float*)alloc(sizeof(float) * 32 * DHEAD * DHEAD);   // contiguous with Ksum
  float*  Ksum = (float*)alloc(sizeof(float) * 32 * DHEAD);
  int*    lens = (int*)alloc(sizeof(int) * NB);
  bf16_t* WqT  = (bf16_t*)alloc(sizeof(bf16_t) * NLAYER * DMODEL * DMODEL);
  bf16_t* WkT  = (bf16_t*)alloc(sizeof(bf16_t) * NLAYER * DMODEL * DMODEL);
  bf16_t* WvT  = (bf16_t*)alloc(sizeof(bf16_t) * NLAYER * DMODEL * DMODEL);
  bf16_t* WoT  = (bf16_t*)alloc(sizeof(bf16_t) * NLAYER * DMODEL * DMODEL);
  bf16_t* W1T  = (bf16_t*)alloc(sizeof(bf16_t) * NLAYER * DMODEL * DFFN);
  bf16_t* W2T  = (bf16_t*)alloc(sizeof(bf16_t) * NLAYER * DMODEL * DFFN);
  float*  Qp   = (float*)alloc(sizeof(float) * NLD);
  float*  Kp   = (float*)alloc(sizeof(float) * NLD);
  float*  Vb   = (float*)alloc(sizeof(float) * NLD);
  float*  Yb   = (float*)alloc(sizeof(float) * NLD);
  bf16_t* Xb   = (bf16_t*)alloc(sizeof(bf16_t) * NLD);
  bf16_t* Ab   = (bf16_t*)alloc(sizeof(bf16_t) * NLD);
  bf16_t* H1b  = (bf16_t*)alloc(sizeof(bf16_t) * (size_t)ROWS * DFFN);
  float*  X    = (float*)d_out;                     // f32 x lives in d_out

  lengths_kernel<<<NB, 256, 0, stream>>>(mask, lens);
  f32_to_bf16_kernel<<<(int)(NLD / 256), 256, 0, stream>>>(src, Xb, (int)NLD);

  for (int i = 0; i < NLAYER; i++) {
    const size_t oDD = (size_t)i * DMODEL * DMODEL;
    const size_t oDF = (size_t)i * DMODEL * DFFN;
    dim3 gT(DMODEL / 32, DMODEL / 32);
    convert_transpose_kernel<<<gT, 256, 0, stream>>>(Wq + oDD, WqT + oDD, DMODEL, DMODEL);
    convert_transpose_kernel<<<gT, 256, 0, stream>>>(Wk + oDD, WkT + oDD, DMODEL, DMODEL);
    convert_transpose_kernel<<<gT, 256, 0, stream>>>(Wv + oDD, WvT + oDD, DMODEL, DMODEL);
    convert_transpose_kernel<<<gT, 256, 0, stream>>>(Wo + oDD, WoT + oDD, DMODEL, DMODEL);
    convert_transpose_kernel<<<dim3(DFFN / 32, DMODEL / 32), 256, 0, stream>>>(
        W1 + oDF, W1T + oDF, DMODEL, DFFN);
    convert_transpose_kernel<<<dim3(DMODEL / 32, DFFN / 32), 256, 0, stream>>>(
        W2 + oDF, W2T + oDF, DFFN, DMODEL);
  }

  const dim3 gD(DMODEL / 128, ROWS / 128);   // (4, 256)
  const dim3 gF(DFFN / 128, ROWS / 128);     // (16, 256)
  const int  znum = 32 * DHEAD * DHEAD + 32 * DHEAD;

  for (int i = 0; i < NLAYER; i++) {
    const float* xf_in = (i == 0) ? src : X;       // f32 residual source
    const size_t oDD = (size_t)i * DMODEL * DMODEL;
    const size_t oDF = (size_t)i * DMODEL * DFFN;
    const size_t oD  = (size_t)i * DMODEL;
    const size_t oF  = (size_t)i * DFFN;

    gemm_bf16_kernel<EPI_ELU1, DMODEL, DMODEL><<<gD, 256, 0, stream>>>(
        Xb, WqT + oDD, bq + oD, nullptr, nullptr, Qp, nullptr);
    gemm_bf16_kernel<EPI_ELU1_MASK, DMODEL, DMODEL><<<gD, 256, 0, stream>>>(
        Xb, WkT + oDD, bk + oD, nullptr, lens, Kp, nullptr);
    gemm_bf16_kernel<EPI_BIAS, DMODEL, DMODEL><<<gD, 256, 0, stream>>>(
        Xb, WvT + oDD, bv + oD, nullptr, nullptr, Vb, nullptr);

    zero_kernel<<<(znum + 255) / 256, 256, 0, stream>>>(KVb, znum);
    kv_reduce_kernel<<<dim3(32, 32), 256, 0, stream>>>(Kp, Vb, KVb, Ksum);
    attn_apply_kernel<<<dim3(32, 64), 256, 0, stream>>>(Qp, KVb, Ksum, Ab);

    gemm_bf16_kernel<EPI_RESID, DMODEL, DMODEL><<<gD, 256, 0, stream>>>(
        Ab, WoT + oDD, bo + oD, xf_in, nullptr, Yb, nullptr);
    ln_kernel<<<ROWS / 8, 256, 0, stream>>>(Yb, X, Xb, g1 + oD, be1 + oD, ROWS);

    gemm_bf16_kernel<EPI_GELU, DMODEL, DFFN><<<gF, 256, 0, stream>>>(
        Xb, W1T + oDF, b1 + oF, nullptr, nullptr, nullptr, H1b);
    gemm_bf16_kernel<EPI_RESID, DFFN, DMODEL><<<gD, 256, 0, stream>>>(
        H1b, W2T + oDF, b2 + oD, X, nullptr, Yb, nullptr);
    ln_kernel<<<ROWS / 8, 256, 0, stream>>>(Yb, X, Xb, g2 + oD, be2 + oD, ROWS);
  }
  ln_kernel<<<ROWS / 8, 256, 0, stream>>>(X, X, nullptr, gf, bef, ROWS);
}
